// SelfAttention_41291815584300
// MI455X (gfx1250) — matrix operand order repaired and checked by reading, not compiled, not run on hardware
//
#include <hip/hip_runtime.h>
#include <hip/hip_bf16.h>

// ---------- types ----------
typedef __bf16 v16bf __attribute__((ext_vector_type(16)));
typedef float  v8f   __attribute__((ext_vector_type(8)));
typedef unsigned short u16;
typedef unsigned int   u32;

union Frag16 { v16bf v; uint4 q[2]; u16 u[16]; };
union FragC  { v8f v; float f[8]; };

__device__ __forceinline__ u16 f2bf(float f) {
  u32 u = __float_as_uint(f);
  u32 r = u + 0x7FFFu + ((u >> 16) & 1u);          // round-to-nearest-even
  if ((u & 0x7F800000u) == 0x7F800000u) r = u;     // inf/nan passthrough
  return (u16)(r >> 16);
}
__device__ __forceinline__ float bf2f(u16 h) {
  return __uint_as_float(((u32)h) << 16);
}
__device__ __forceinline__ v8f wmma_bf16(v16bf a, v16bf b, v8f c) {
  // D = A(16x32 bf16) * B(32x16 bf16) + C(16x16 f32)
  return __builtin_amdgcn_wmma_f32_16x16x32_bf16(false, a, false, b,
                                                 (short)0, c, false, false);
}

// ---------- 1) f32 -> bf16 convert (vectorized) ----------
__global__ void k_cvt_bf16(const float* __restrict__ src, u16* __restrict__ dst, int n4) {
  int i = blockIdx.x * blockDim.x + threadIdx.x;
  if (i >= n4) return;
  float4 f = ((const float4*)src)[i];
  ushort4 o;
  o.x = f2bf(f.x); o.y = f2bf(f.y); o.z = f2bf(f.z); o.w = f2bf(f.w);
  ((ushort4*)dst)[i] = o;
}

// ---------- 2) bf16 WMMA GEMM: C[M,N] = A[M,K] @ B[K,N] ----------
// Block tile 128x128, K-step 32, 8 waves: each wave computes 32x64 (2x4 subtiles).
// Double-buffered LDS: global loads for tile t+1 issued before computing tile t.
#define GBM 128
#define GBN 128
#define GBK 32
#define GPAD 8

template <int STORE_F32>
__global__ __launch_bounds__(256, 1) void k_gemm_bf16(
    const u16* __restrict__ A, const u16* __restrict__ B,
    u16* __restrict__ Cb, float* __restrict__ Cf,
    int N, int K) {
  __shared__ u16 As [2][GBM][GBK + GPAD];   // [buf][m][k]
  __shared__ u16 Bst[2][GBN][GBK + GPAD];   // [buf][n][k] transposed
  const int tid  = threadIdx.x;
  const int lane = tid & 31;
  const int w    = tid >> 5;
  const int wm   = w & 3;       // 4 row groups of 32
  const int wn   = w >> 2;      // 2 col groups of 64
  const int c    = lane & 15;
  const int hh   = lane >> 4;
  const int rowBase = blockIdx.y * GBM;
  const int colBase = blockIdx.x * GBN;

  const int arow = tid >> 1, akg = tid & 1;   // A coop load: 16 elems/thread
  const int bk   = tid >> 3, bng = tid & 7;   // B coop load: 16 elems/thread
  const u16* aptr = A + (size_t)(rowBase + arow) * K + akg * 16;
  const u16* bptr = B + (size_t)bk * N + colBase + bng * 16;
  const size_t bstride = (size_t)GBK * N;

  FragC acc[2][4];
#pragma unroll
  for (int mi = 0; mi < 2; mi++)
#pragma unroll
    for (int ni = 0; ni < 4; ni++)
#pragma unroll
      for (int e = 0; e < 8; e++) acc[mi][ni].f[e] = 0.f;

  uint4 ra0, ra1, rb0, rb1;
  // prologue: stage tile 0
  ra0 = ((const uint4*)aptr)[0];
  ra1 = ((const uint4*)aptr)[1];
  rb0 = ((const uint4*)bptr)[0];
  rb1 = ((const uint4*)bptr)[1];
  *((uint4*)&As[0][arow][akg * 16 + 0]) = ra0;
  *((uint4*)&As[0][arow][akg * 16 + 8]) = ra1;
  {
    u16 tmp[16];
    __builtin_memcpy(tmp,     &rb0, 16);
    __builtin_memcpy(tmp + 8, &rb1, 16);
#pragma unroll
    for (int j = 0; j < 16; j++) Bst[0][bng * 16 + j][bk] = tmp[j];
  }
  __syncthreads();

  const int nk = K / GBK;
#pragma unroll 1
  for (int kt = 0; kt < nk; kt++) {
    const int cur = kt & 1;
    const bool more = (kt + 1) < nk;
    if (more) {  // issue global loads for next tile (latency hidden by WMMAs)
      const u16* an = aptr + (kt + 1) * GBK;
      const u16* bn = bptr + (size_t)(kt + 1) * bstride;
      ra0 = ((const uint4*)an)[0];
      ra1 = ((const uint4*)an)[1];
      rb0 = ((const uint4*)bn)[0];
      rb1 = ((const uint4*)bn)[1];
    }

    // compute tile kt from LDS[cur]
    Frag16 af[2];
#pragma unroll
    for (int mi = 0; mi < 2; mi++) {
      int r = wm * 32 + mi * 16 + c;                      // M row of this lane
      af[mi].q[0] = *((const uint4*)&As[cur][r][hh * 8]);       // K = h*8..h*8+7
      af[mi].q[1] = *((const uint4*)&As[cur][r][16 + hh * 8]);  // K = 16+h*8..
    }
#pragma unroll
    for (int ni = 0; ni < 4; ni++) {
      Frag16 bf;
      int n = wn * 64 + ni * 16 + c;                      // N col of this lane
      bf.q[0] = *((const uint4*)&Bst[cur][n][hh * 8]);
      bf.q[1] = *((const uint4*)&Bst[cur][n][16 + hh * 8]);
#pragma unroll
      for (int mi = 0; mi < 2; mi++)
        acc[mi][ni].v = wmma_bf16(af[mi].v, bf.v, acc[mi][ni].v);
    }

    if (more) {  // stage next tile into the other buffer
      const int nxt = cur ^ 1;
      *((uint4*)&As[nxt][arow][akg * 16 + 0]) = ra0;
      *((uint4*)&As[nxt][arow][akg * 16 + 8]) = ra1;
      u16 tmp[16];
      __builtin_memcpy(tmp,     &rb0, 16);
      __builtin_memcpy(tmp + 8, &rb1, 16);
#pragma unroll
      for (int j = 0; j < 16; j++) Bst[nxt][bng * 16 + j][bk] = tmp[j];
    }
    __syncthreads();
  }

  // straight-line epilogue (compile-time output type)
#pragma unroll
  for (int mi = 0; mi < 2; mi++)
#pragma unroll
    for (int ni = 0; ni < 4; ni++) {
      int col = colBase + wn * 64 + ni * 16 + c;
#pragma unroll
      for (int vv = 0; vv < 8; vv++) {
        int row = rowBase + wm * 32 + mi * 16 + vv + 8 * hh;
        if (STORE_F32) Cf[(size_t)row * N + col] = acc[mi][ni].f[vv];
        else           Cb[(size_t)row * N + col] = f2bf(acc[mi][ni].f[vv]);
      }
    }
}

// ---------- 3) RoPE + scatter qkv[4096,6144] -> q/k/v [b,h,l,d] ----------
__global__ void k_rope(const u16* __restrict__ qkv, u16* __restrict__ q,
                       u16* __restrict__ kk, u16* __restrict__ v) {
  int idx = blockIdx.x * blockDim.x + threadIdx.x;   // b*l*h*64
  int i = idx & 63;
  int h = (idx >> 6) & 15;
  int l = (idx >> 10) & 2047;
  int b = idx >> 21;
  float inv = __powf(10000.0f, -(float)(2 * i) * (1.0f / 128.0f));
  float ang = (float)l * inv;
  float s, cv; __sincosf(ang, &s, &cv);
  size_t row  = (size_t)(b * 2048 + l) * 6144;
  size_t orow = ((size_t)((b * 16 + h) * 2048 + l)) * 128;
  float q1 = bf2f(qkv[row + h * 128 + i]);
  float q2 = bf2f(qkv[row + h * 128 + i + 64]);
  q[orow + i]      = f2bf(q1 * cv - q2 * s);
  q[orow + i + 64] = f2bf(q2 * cv + q1 * s);
  float k1 = bf2f(qkv[row + 2048 + h * 128 + i]);
  float k2 = bf2f(qkv[row + 2048 + h * 128 + i + 64]);
  kk[orow + i]      = f2bf(k1 * cv - k2 * s);
  kk[orow + i + 64] = f2bf(k2 * cv + k1 * s);
  v[orow + i]      = qkv[row + 4096 + h * 128 + i];
  v[orow + i + 64] = qkv[row + 4096 + h * 128 + i + 64];
}

// ---------- 4) sigmoid gate: g[row,h] = sigmoid(x @ Wg + bg) ----------
__global__ __launch_bounds__(256) void k_gate(const float* __restrict__ x,
                                              const float* __restrict__ Wg,
                                              const float* __restrict__ bg,
                                              float* __restrict__ gate) {
  int w = threadIdx.x >> 5, lane = threadIdx.x & 31;
  int row = blockIdx.x * 8 + w;
  float acc[16];
#pragma unroll
  for (int h = 0; h < 16; h++) acc[h] = 0.f;
  for (int k = lane; k < 2048; k += 32) {
    float xv = x[(size_t)row * 2048 + k];
#pragma unroll
    for (int h = 0; h < 16; h++) acc[h] += xv * Wg[k * 16 + h];
  }
#pragma unroll
  for (int h = 0; h < 16; h++)
#pragma unroll
    for (int st = 1; st < 32; st <<= 1) acc[h] += __shfl_xor(acc[h], st, 32);
  if (lane < 16) {
    float g = acc[lane] + bg[lane];
    gate[(size_t)row * 16 + lane] = 1.0f / (1.0f + __expf(-g));
  }
}

// ---------- 5) flash attention + gate, bf16 WMMA ----------
// Block = (q-tile of 128 rows) x (b,h). 8 waves, 16 q rows each. BK=64 keys/tile.
#define FBK 64
__global__ __launch_bounds__(256, 1) void k_flash(
    const u16* __restrict__ q, const u16* __restrict__ k,
    const u16* __restrict__ v, const float* __restrict__ gate,
    u16* __restrict__ attb) {
  __shared__ u16 Ks[FBK][128 + 8];     // [key][d]  (B frag of Q*K^T contiguous in d)
  __shared__ u16 Vt[128][FBK + 8];     // [d][key]  (B frag of P*V contiguous in key)
  __shared__ u16 Ps[8][16][FBK + 8];   // per-wave P strip (A operand of P*V)
  const int tid  = threadIdx.x;
  const int lane = tid & 31;
  const int w    = tid >> 5;
  const int c    = lane & 15;
  const int hh   = lane >> 4;
  const int qtile = blockIdx.x;        // 0..15
  const int bh    = blockIdx.y;        // 0..31
  const int b = bh >> 4, head = bh & 15;
  const size_t seqbase = (size_t)bh * 2048;
  const int q0 = qtile * 128;
  const int qrow_w = q0 + w * 16;

  // Q fragments for this wave's 16 rows (4 d-chunks of 32), loaded once
  Frag16 qf[4];
  {
    const u16* qr = q + (seqbase + qrow_w + c) * 128;
#pragma unroll
    for (int dc = 0; dc < 4; dc++) {
      qf[dc].q[0] = *((const uint4*)(qr + dc * 32 + hh * 8));
      qf[dc].q[1] = *((const uint4*)(qr + dc * 32 + 16 + hh * 8));
    }
  }

  FragC oacc[8];
#pragma unroll
  for (int td = 0; td < 8; td++)
#pragma unroll
    for (int e = 0; e < 8; e++) oacc[td].f[e] = 0.f;
  float mrow[8], lrow[8];
#pragma unroll
  for (int vv = 0; vv < 8; vv++) { mrow[vv] = -3.0e38f; lrow[vv] = 0.f; }

  const float scale = 0.08838834764831845f;   // 1/sqrt(128)
  const int nkt = (q0 + 128) / FBK;           // causal: only keys <= last q row
  const int crow = tid >> 2, cdg = tid & 3;   // coop-load decomposition

#pragma unroll 1
  for (int kt = 0; kt < nkt; kt++) {
    const int kbase = kt * FBK;
    { // load K tile row-major, V tile transposed
      const u16* ksrc = k + (seqbase + kbase + crow) * 128 + cdg * 32;
#pragma unroll
      for (int j = 0; j < 4; j++)
        *((uint4*)&Ks[crow][cdg * 32 + j * 8]) = ((const uint4*)ksrc)[j];
      const u16* vsrc = v + (seqbase + kbase + crow) * 128 + cdg * 32;
#pragma unroll
      for (int j = 0; j < 4; j++) {
        uint4 val = ((const uint4*)vsrc)[j];
        u16 tmp[8]; __builtin_memcpy(tmp, &val, 16);
#pragma unroll
        for (int e = 0; e < 8; e++) Vt[cdg * 32 + j * 8 + e][crow] = tmp[e];
      }
    }
    __syncthreads();

    // S strip = Q(16x128) * K^T(128x64): 4 n-subtiles x 4 d-chunks
    FragC sacc[4];
#pragma unroll
    for (int t = 0; t < 4; t++) {
#pragma unroll
      for (int e = 0; e < 8; e++) sacc[t].f[e] = 0.f;
#pragma unroll
      for (int dc = 0; dc < 4; dc++) {
        Frag16 bf;
        bf.q[0] = *((const uint4*)&Ks[t * 16 + c][dc * 32 + hh * 8]);
        bf.q[1] = *((const uint4*)&Ks[t * 16 + c][dc * 32 + 16 + hh * 8]);
        sacc[t].v = wmma_bf16(qf[dc].v, bf.v, sacc[t].v);
      }
    }
    // scale + causal mask
#pragma unroll
    for (int t = 0; t < 4; t++) {
      int kcol = kbase + t * 16 + c;
#pragma unroll
      for (int vv = 0; vv < 8; vv++) {
        float s = sacc[t].f[vv] * scale;
        if (kcol > qrow_w + vv + 8 * hh) s = -3.0e38f;
        sacc[t].f[vv] = s;
      }
    }
    // online softmax (row lives in lanes of one 16-lane half -> xor reduce 1,2,4,8)
    float alpha[8];
#pragma unroll
    for (int vv = 0; vv < 8; vv++) {
      float m = sacc[0].f[vv];
#pragma unroll
      for (int t = 1; t < 4; t++) m = fmaxf(m, sacc[t].f[vv]);
#pragma unroll
      for (int st = 1; st < 16; st <<= 1) m = fmaxf(m, __shfl_xor(m, st, 32));
      float mn = fmaxf(mrow[vv], m);
      alpha[vv] = __expf(mrow[vv] - mn);
      mrow[vv] = mn;
      float rs = 0.f;
#pragma unroll
      for (int t = 0; t < 4; t++) {
        float p = __expf(sacc[t].f[vv] - mn);
        rs += p;
        Ps[w][vv + 8 * hh][t * 16 + c] = f2bf(p);
      }
#pragma unroll
      for (int st = 1; st < 16; st <<= 1) rs += __shfl_xor(rs, st, 32);
      lrow[vv] = lrow[vv] * alpha[vv] + rs;
#pragma unroll
      for (int td = 0; td < 8; td++) oacc[td].f[vv] *= alpha[vv];
    }
    __syncthreads();

    // O += P(16x64) * V(64x128): 2 k-chunks x 8 d-subtiles
#pragma unroll
    for (int cc = 0; cc < 2; cc++) {
      Frag16 pa;
      pa.q[0] = *((const uint4*)&Ps[w][c][cc * 32 + hh * 8 + 0]);
      pa.q[1] = *((const uint4*)&Ps[w][c][cc * 32 + 16 + hh * 8]);
#pragma unroll
      for (int td = 0; td < 8; td++) {
        Frag16 bf;
        bf.q[0] = *((const uint4*)&Vt[td * 16 + c][cc * 32 + hh * 8]);
        bf.q[1] = *((const uint4*)&Vt[td * 16 + c][cc * 32 + 16 + hh * 8]);
        oacc[td].v = wmma_bf16(pa.v, bf.v, oacc[td].v);
      }
    }
    __syncthreads();
  }

  // epilogue: normalize by row-sum, apply sigmoid gate, write [row, h*128+d] bf16
  float gv[8];
#pragma unroll
  for (int vv = 0; vv < 8; vv++) {
    int qr = qrow_w + vv + 8 * hh;
    gv[vv] = gate[(size_t)(b * 2048 + qr) * 16 + head] / lrow[vv];
  }
#pragma unroll
  for (int td = 0; td < 8; td++)
#pragma unroll
    for (int vv = 0; vv < 8; vv++) {
      int qr = qrow_w + vv + 8 * hh;
      size_t o = (size_t)(b * 2048 + qr) * 2048 + head * 128 + td * 16 + c;
      attb[o] = f2bf(oacc[td].f[vv] * gv[vv]);
    }
}

// ---------- launch ----------
extern "C" void kernel_launch(void* const* d_in, const int* in_sizes, int n_in,
                              void* d_out, int out_size, void* d_ws, size_t ws_size,
                              hipStream_t stream) {
  const float* x    = (const float*)d_in[0];   // [2,2048,2048]
  const float* Wqkv = (const float*)d_in[1];   // [2048,6144]
  const float* Wg   = (const float*)d_in[2];   // [2048,16]
  const float* bg   = (const float*)d_in[3];   // [16]
  const float* Wfc  = (const float*)d_in[4];   // [2048,2048]
  float* out = (float*)d_out;                  // [2,2048,2048] f32
  char* ws = (char*)d_ws;

  const size_t BL = 4096, HID = 2048, NQKV = 6144;
  u16* xb      = (u16*)(ws + 0);               // 16 MB
  u16* wqkvb   = (u16*)(ws + 16777216);        // 24 MB
  u16* wfcb    = (u16*)(ws + 41943040);        // 8 MB
  u16* qkvb    = (u16*)(ws + 50331648);        // 48 MB
  u16* qb      = (u16*)(ws + 100663296);       // 16 MB
  u16* kb      = (u16*)(ws + 117440512);       // 16 MB
  u16* vb      = (u16*)(ws + 134217728);       // 16 MB
  float* gateb = (float*)(ws + 150994944);     // 256 KB
  u16* attb    = (u16*)(ws + 151257088);       // 16 MB  (total ~168 MB)

  k_cvt_bf16<<<(int)((BL * HID / 4 + 255) / 256), 256, 0, stream>>>(x, xb, (int)(BL * HID / 4));
  k_cvt_bf16<<<(int)((HID * NQKV / 4 + 255) / 256), 256, 0, stream>>>(Wqkv, wqkvb, (int)(HID * NQKV / 4));
  k_cvt_bf16<<<(int)((HID * HID / 4 + 255) / 256), 256, 0, stream>>>(Wfc, wfcb, (int)(HID * HID / 4));

  // QKV projection (bf16 out)
  k_gemm_bf16<0><<<dim3(NQKV / 128, BL / 128), 256, 0, stream>>>(
      xb, wqkvb, qkvb, nullptr, (int)NQKV, (int)HID);
  // RoPE + head-major scatter
  k_rope<<<(2 * 2048 * 16 * 64) / 256, 256, 0, stream>>>(qkvb, qb, kb, vb);
  // gate
  k_gate<<<4096 / 8, 256, 0, stream>>>(x, Wg, bg, gateb);
  // flash attention + gate -> attb [4096, 2048] bf16
  k_flash<<<dim3(16, 32), 256, 0, stream>>>(qb, kb, vb, gateb, attb);
  // output projection (f32 out)
  k_gemm_bf16<1><<<dim3(HID / 128, BL / 128), 256, 0, stream>>>(
      attb, wfcb, nullptr, out, (int)HID, (int)HID);
}
